// SimpleMultiheadAttention_44727789420890
// MI455X (gfx1250) — matrix operand order repaired and checked by reading, not compiled, not run on hardware
//
#include <hip/hip_runtime.h>
#include <hip/hip_bf16.h>
#include <stdint.h>

// ---------------- problem constants ----------------
#define D_EMBED 1024
#define N_SEQ   4096
#define B_BATCH 4
#define H_HEADS 16
#define DH      64
#define M_ROWS  (B_BATCH * N_SEQ)   // 16384
#define SCALE_F 0.125f              // 64^-0.5

typedef __bf16 bf16_t;
typedef __attribute__((ext_vector_type(8)))  __bf16 v8bf;
typedef __attribute__((ext_vector_type(16))) __bf16 v16bf;
typedef __attribute__((ext_vector_type(8)))  float  v8f;
typedef __attribute__((ext_vector_type(4)))  unsigned int v4u;
typedef __attribute__((ext_vector_type(8)))  int v8i_t;
typedef __attribute__((ext_vector_type(4)))  int v4i_t;

// LDS tile geometry: 64 bf16 per row (128 B) + 16 B TDM pad -> 144 B row stride
// (bank stride of 36 on the 64-bank LDS => conflict-free fragment reads).
#define KSTEP       64
#define LDS_ROW_BF  72          // 144 bytes / 2

__device__ __forceinline__ bf16_t f2bf(float f) {
    union { float f; uint32_t u; } c; c.f = f;
    uint32_t r = c.u + 0x7FFFu + ((c.u >> 16) & 1u);   // round-to-nearest-even
    union { uint16_t u; bf16_t b; } o; o.u = (uint16_t)(r >> 16);
    return o.b;
}

// ---------------- f32 -> bf16 conversion (memory bound) ----------------
__global__ __launch_bounds__(256) void cvt_f32_bf16_kernel(
    const float* __restrict__ src, bf16_t* __restrict__ dst, int n)
{
    int i = (blockIdx.x * 256 + threadIdx.x) * 4;
    if (i + 3 < n) {
        float4 v = *reinterpret_cast<const float4*>(src + i);
        bf16_t o[4] = { f2bf(v.x), f2bf(v.y), f2bf(v.z), f2bf(v.w) };
        *reinterpret_cast<uint64_t*>(dst + i) = *reinterpret_cast<uint64_t*>(o);
    }
}

// ---------------- TDM tile load: 128 rows x 64 bf16, row stride K, padded into LDS ----
// D# packing per CDNA5 ISA:
//  g0: [1:0]=count=1 | [63:32]=lds_addr | [120:64]=global_addr | [127:126]=type=2
//  g1: data_size=1(2B), pad_enable, pad_interval=4 (32 DW), pad_amount enc 3 (4 DW),
//      tensor_dim0=K, tensor_dim1=rows, tile_dim0=64, tile_dim1=128, dim0_stride=K
// This toolchain exposes the 6-arg builtin:
//   (uint32x4 g0, int32x8 g1, int32x4 g2, int32x4 g3, int32x8 extra, i32 cpol)
__device__ __forceinline__ void tdm_load_tile(const bf16_t* g, uint32_t lds_byte,
                                              uint32_t K, uint32_t tensor_rows)
{
    uint64_t ga = (uint64_t)(uintptr_t)g;
    v4u g0;
    g0[0] = 1u;                                    // count=1, is_restore=0, gather off
    g0[1] = lds_byte;                              // lds_addr
    g0[2] = (uint32_t)ga;                          // global_addr[31:0]
    g0[3] = (uint32_t)(ga >> 32) | 0x80000000u;    // global_addr[56:32] | type=2
    v8i_t g1;
    g1[0] = (int)((1u << 16) | (1u << 20) | (4u << 22) | (3u << 25)); // dsz=2B, pad 4DW/32DW
    g1[1] = (int)(K << 16);                        // tensor_dim0[15:0] -> bits [63:48]
    g1[2] = (int)((K >> 16) | (tensor_rows << 16));// tensor_dim0 hi | tensor_dim1 lo
    g1[3] = (int)((tensor_rows >> 16) | (64u << 16)); // tensor_dim1 hi | tile_dim0=64
    g1[4] = 128;                                   // tile_dim1=128, tile_dim2=0
    g1[5] = (int)K;                                // tensor_dim0_stride[31:0]
    g1[6] = 0;                                     // stride hi | dim1_stride lo
    g1[7] = 0;                                     // dim1_stride hi
    v4i_t z4; z4[0] = 0; z4[1] = 0; z4[2] = 0; z4[3] = 0;
    v8i_t z8;
#pragma unroll
    for (int i = 0; i < 8; ++i) z8[i] = 0;
    __builtin_amdgcn_tensor_load_to_lds(g0, g1, z4, z4, z8, 0);
}

// ---------------- bf16 WMMA GEMM (TDM-staged):  C = A[M,K] * B[N,K]^T + bias ----------
// Block: 256 threads = 8 waves. Block tile 128(M) x 128(N), K-step 64, LDS double-buffer.
// Waves: 4 along M (32 rows) x 2 along N (64 cols); wave tile 32x64 = 2x4 fragments.
__global__ __launch_bounds__(256) void gemm_bf16_tdm(
    const bf16_t* __restrict__ A,   // [M,K] row-major
    const bf16_t* __restrict__ Bw,  // [N,K] row-major
    const float*  __restrict__ bias,// [N]
    float* __restrict__ C,          // [M,N] row-major f32
    int M, int N, int K)
{
    __shared__ bf16_t Alds[2][128 * LDS_ROW_BF];   // 2 x 18 KB
    __shared__ bf16_t Blds[2][128 * LDS_ROW_BF];   // 2 x 18 KB

    const int lane = threadIdx.x & 31;
    const int wave = threadIdx.x >> 5;
    const int half = lane >> 4;
    const int l16  = lane & 15;
    const int wm   = wave & 3;
    const int wn   = wave >> 2;
    const int mblk = blockIdx.y * 128;
    const int nblk = blockIdx.x * 128;
    const int n0   = nblk + wn * 64;

    // Accumulators initialized with bias.
    v8f acc[2][4];
#pragma unroll
    for (int mi = 0; mi < 2; ++mi)
#pragma unroll
        for (int ni = 0; ni < 4; ++ni) {
            float bn = bias[n0 + ni * 16 + l16];
#pragma unroll
            for (int r = 0; r < 8; ++r) acc[mi][ni][r] = bn;
        }

    const uint32_t aoff0 = (uint32_t)(uintptr_t)&Alds[0][0];
    const uint32_t aoff1 = (uint32_t)(uintptr_t)&Alds[1][0];
    const uint32_t boff0 = (uint32_t)(uintptr_t)&Blds[0][0];
    const uint32_t boff1 = (uint32_t)(uintptr_t)&Blds[1][0];

    // Prologue: fetch first K-slab into buffer 0 (wave0: A tile, wave1: B tile).
    if (wave == 0)      tdm_load_tile(A  + (size_t)mblk * K, aoff0, (uint32_t)K, (uint32_t)M);
    else if (wave == 1) tdm_load_tile(Bw + (size_t)nblk * K, boff0, (uint32_t)K, (uint32_t)N);
    __builtin_amdgcn_s_wait_tensorcnt(0);
    __syncthreads();

    // Per-lane LDS fragment bases (padded row stride).
    // A fragment: lane l%16 = row; lanes 0-15 hold K[0..7],K[16..23]; 16-31 hold K[8..15],K[24..31].
    const int arow0 = (wm * 32 +  0 + l16) * LDS_ROW_BF + half * 8;
    const int arow1 = (wm * 32 + 16 + l16) * LDS_ROW_BF + half * 8;
    // B fragment: lane l%16 = output col; lanes 0-15 hold K[0..7],K[16..23]; 16-31 hold K[8..15],K[24..31].
    int brow[4];
#pragma unroll
    for (int ni = 0; ni < 4; ++ni)
        brow[ni] = (wn * 64 + ni * 16 + l16) * LDS_ROW_BF + half * 8;

    int buf = 0;
    for (int k0 = 0; k0 < K; k0 += KSTEP) {
        // Prefetch next slab into the other buffer while computing on this one.
        if (k0 + KSTEP < K) {
            if (wave == 0)
                tdm_load_tile(A + (size_t)mblk * K + (k0 + KSTEP),
                              buf ? aoff0 : aoff1, (uint32_t)K, (uint32_t)M);
            else if (wave == 1)
                tdm_load_tile(Bw + (size_t)nblk * K + (k0 + KSTEP),
                              buf ? boff0 : boff1, (uint32_t)K, (uint32_t)N);
        }
        const bf16_t* Ab = Alds[buf];
        const bf16_t* Bb = Blds[buf];
#pragma unroll
        for (int kk = 0; kk < KSTEP; kk += 32) {
            v16bf a[2], b[4];
            {
                v8bf lo0 = *reinterpret_cast<const v8bf*>(Ab + arow0 + kk);
                v8bf hi0 = *reinterpret_cast<const v8bf*>(Ab + arow0 + kk + 16);
                a[0] = __builtin_shufflevector(lo0, hi0, 0,1,2,3,4,5,6,7,8,9,10,11,12,13,14,15);
                v8bf lo1 = *reinterpret_cast<const v8bf*>(Ab + arow1 + kk);
                v8bf hi1 = *reinterpret_cast<const v8bf*>(Ab + arow1 + kk + 16);
                a[1] = __builtin_shufflevector(lo1, hi1, 0,1,2,3,4,5,6,7,8,9,10,11,12,13,14,15);
            }
#pragma unroll
            for (int ni = 0; ni < 4; ++ni) {
                v8bf blo = *reinterpret_cast<const v8bf*>(Bb + brow[ni] + kk);
                v8bf bhi = *reinterpret_cast<const v8bf*>(Bb + brow[ni] + kk + 16);
                b[ni] = __builtin_shufflevector(blo, bhi, 0,1,2,3,4,5,6,7,8,9,10,11,12,13,14,15);
            }
#pragma unroll
            for (int mi = 0; mi < 2; ++mi)
#pragma unroll
                for (int ni = 0; ni < 4; ++ni)
                    acc[mi][ni] = __builtin_amdgcn_wmma_f32_16x16x32_bf16(
                        false, a[mi], false, b[ni], (short)0, acc[mi][ni], false, false);
        }
        __builtin_amdgcn_s_wait_tensorcnt(0);
        __syncthreads();
        buf ^= 1;
    }

    // Store: fragment element r -> row mblk + wm*32 + mi*16 + half*8 + r, col n0 + ni*16 + l16.
#pragma unroll
    for (int mi = 0; mi < 2; ++mi)
#pragma unroll
        for (int ni = 0; ni < 4; ++ni) {
            int n = n0 + ni * 16 + l16;
#pragma unroll
            for (int r = 0; r < 8; ++r) {
                int m = mblk + wm * 32 + mi * 16 + half * 8 + r;
                C[(size_t)m * N + n] = acc[mi][ni][r];
            }
        }
}

// ---------------- pass 2: alpha softmax + global_q, one block per (b,h) ----------------
__global__ __launch_bounds__(256) void global_q_kernel(
    const float* __restrict__ Qf,      // [M_ROWS, D_EMBED] f32
    const float* __restrict__ q_alpha, // [H, DH]
    float* __restrict__ gq)            // [B*H, DH]
{
    __shared__ float wbuf[N_SEQ];
    __shared__ float red[8];
    __shared__ float sqa[DH];
    __shared__ float p2[256];

    const int b = blockIdx.x / H_HEADS, h = blockIdx.x % H_HEADS;
    const int tid = threadIdx.x, lane = tid & 31, wave = tid >> 5;
    const float* Qb = Qf + (size_t)b * N_SEQ * D_EMBED + h * DH;

    if (tid < DH) sqa[tid] = q_alpha[h * DH + tid];
    __syncthreads();

    float lmax = -3.0e38f;
    for (int n = tid; n < N_SEQ; n += 256) {
        const float* row = Qb + (size_t)n * D_EMBED;
        float s = 0.f;
#pragma unroll
        for (int d = 0; d < DH; d += 4) {
            float4 qv = *reinterpret_cast<const float4*>(row + d);
            s += qv.x * sqa[d] + qv.y * sqa[d + 1] + qv.z * sqa[d + 2] + qv.w * sqa[d + 3];
        }
        s *= SCALE_F;
        wbuf[n] = s;
        lmax = fmaxf(lmax, s);
    }
#pragma unroll
    for (int off = 16; off; off >>= 1) lmax = fmaxf(lmax, __shfl_xor(lmax, off, 32));
    if (lane == 0) red[wave] = lmax;
    __syncthreads();
    float bmax = red[0];
#pragma unroll
    for (int i = 1; i < 8; ++i) bmax = fmaxf(bmax, red[i]);
    __syncthreads();

    float lsum = 0.f;
    for (int n = tid; n < N_SEQ; n += 256) {
        float e = __expf(wbuf[n] - bmax);
        wbuf[n] = e;
        lsum += e;
    }
#pragma unroll
    for (int off = 16; off; off >>= 1) lsum += __shfl_xor(lsum, off, 32);
    if (lane == 0) red[wave] = lsum;
    __syncthreads();
    float bsum = 0.f;
#pragma unroll
    for (int i = 0; i < 8; ++i) bsum += red[i];
    const float inv = 1.f / bsum;

    const int d = tid & 63, g = tid >> 6;
    float acc = 0.f;
    for (int n = g; n < N_SEQ; n += 4)
        acc += Qb[(size_t)n * D_EMBED + d] * wbuf[n];
    p2[tid] = acc;
    __syncthreads();
    if (tid < 64) {
        float s = p2[tid] + p2[tid + 64] + p2[tid + 128] + p2[tid + 192];
        gq[(size_t)blockIdx.x * DH + tid] = s * inv;
    }
}

// ---------------- pass 3: beta softmax + global_v + fused bf16(Q + global_v) ----------------
__global__ __launch_bounds__(256) void global_v_fuse_kernel(
    const float* __restrict__ Qf, const float* __restrict__ Kf, const float* __restrict__ Vf,
    const float* __restrict__ gq, const float* __restrict__ k_beta,
    bf16_t* __restrict__ Abf)          // [M_ROWS, D_EMBED] bf16 output (Q + global_v)
{
    __shared__ float wbuf[N_SEQ];
    __shared__ float red[8];
    __shared__ float sw[DH];
    __shared__ float sgv[DH];
    __shared__ float p2[256];

    const int b = blockIdx.x / H_HEADS, h = blockIdx.x % H_HEADS;
    const int tid = threadIdx.x, lane = tid & 31, wave = tid >> 5;
    const size_t base = (size_t)b * N_SEQ * D_EMBED + h * DH;
    const float* Kb = Kf + base;
    const float* Vb = Vf + base;

    if (tid < DH) sw[tid] = gq[(size_t)blockIdx.x * DH + tid] * k_beta[h * DH + tid];
    __syncthreads();

    float lmax = -3.0e38f;
    for (int n = tid; n < N_SEQ; n += 256) {
        const float* row = Kb + (size_t)n * D_EMBED;
        float s = 0.f;
#pragma unroll
        for (int d = 0; d < DH; d += 4) {
            float4 kv = *reinterpret_cast<const float4*>(row + d);
            s += kv.x * sw[d] + kv.y * sw[d + 1] + kv.z * sw[d + 2] + kv.w * sw[d + 3];
        }
        s *= SCALE_F;
        wbuf[n] = s;
        lmax = fmaxf(lmax, s);
    }
#pragma unroll
    for (int off = 16; off; off >>= 1) lmax = fmaxf(lmax, __shfl_xor(lmax, off, 32));
    if (lane == 0) red[wave] = lmax;
    __syncthreads();
    float bmax = red[0];
#pragma unroll
    for (int i = 1; i < 8; ++i) bmax = fmaxf(bmax, red[i]);
    __syncthreads();

    float lsum = 0.f;
    for (int n = tid; n < N_SEQ; n += 256) {
        float e = __expf(wbuf[n] - bmax);
        wbuf[n] = e;
        lsum += e;
    }
#pragma unroll
    for (int off = 16; off; off >>= 1) lsum += __shfl_xor(lsum, off, 32);
    if (lane == 0) red[wave] = lsum;
    __syncthreads();
    float bsum = 0.f;
#pragma unroll
    for (int i = 0; i < 8; ++i) bsum += red[i];
    const float inv = 1.f / bsum;

    const int d = tid & 63, g = tid >> 6;
    float acc = 0.f;
    for (int n = g; n < N_SEQ; n += 4)
        acc += Vb[(size_t)n * D_EMBED + d] * wbuf[n];
    p2[tid] = acc;
    __syncthreads();
    if (tid < 64)
        sgv[tid] = (p2[tid] + p2[tid + 64] + p2[tid + 128] + p2[tid + 192]) * inv;
    __syncthreads();

    for (int idx = tid; idx < N_SEQ * DH; idx += 256) {
        int n = idx >> 6, dh = idx & 63;
        size_t off = base + (size_t)n * D_EMBED + dh;
        Abf[off] = f2bf(Qf[off] + sgv[dh]);
    }
}

// ---------------- host-side orchestration ----------------
extern "C" void kernel_launch(void* const* d_in, const int* in_sizes, int n_in,
                              void* d_out, int out_size, void* d_ws, size_t ws_size,
                              hipStream_t stream)
{
    const float* q       = (const float*)d_in[0];
    const float* k       = (const float*)d_in[1];
    const float* v       = (const float*)d_in[2];
    const float* Wq      = (const float*)d_in[3];
    const float* bq      = (const float*)d_in[4];
    const float* Wk      = (const float*)d_in[5];
    const float* bk      = (const float*)d_in[6];
    const float* Wv      = (const float*)d_in[7];
    const float* bv      = (const float*)d_in[8];
    const float* q_alpha = (const float*)d_in[9];
    const float* k_beta  = (const float*)d_in[10];
    const float* Wo      = (const float*)d_in[11];
    const float* bo      = (const float*)d_in[12];

    uint8_t* ws = (uint8_t*)d_ws;
    size_t off = 0;
    auto alloc = [&](size_t bytes) { void* p = ws + off; off += (bytes + 255) & ~size_t(255); return p; };
    float*  Qf    = (float*) alloc((size_t)M_ROWS * D_EMBED * 4);
    float*  Kf    = (float*) alloc((size_t)M_ROWS * D_EMBED * 4);
    float*  Vf    = (float*) alloc((size_t)M_ROWS * D_EMBED * 4);
    bf16_t* actbf = (bf16_t*)alloc((size_t)M_ROWS * D_EMBED * 2);   // reused 4x
    bf16_t* Wqbf  = (bf16_t*)alloc((size_t)D_EMBED * D_EMBED * 2);
    bf16_t* Wkbf  = (bf16_t*)alloc((size_t)D_EMBED * D_EMBED * 2);
    bf16_t* Wvbf  = (bf16_t*)alloc((size_t)D_EMBED * D_EMBED * 2);
    bf16_t* Wobf  = (bf16_t*)alloc((size_t)D_EMBED * D_EMBED * 2);
    float*  gq    = (float*) alloc((size_t)B_BATCH * H_HEADS * DH * 4);
    (void)ws_size; (void)in_sizes; (void)n_in; (void)out_size;

    auto cvt = [&](const float* s, bf16_t* d, int n) {
        cvt_f32_bf16_kernel<<<(n / 4 + 255) / 256, 256, 0, stream>>>(s, d, n);
    };

    const int NW = D_EMBED * D_EMBED;
    cvt(Wq, Wqbf, NW);
    cvt(Wk, Wkbf, NW);
    cvt(Wv, Wvbf, NW);
    cvt(Wo, Wobf, NW);

    const dim3 gemm_grid(D_EMBED / 128, M_ROWS / 128, 1);   // (8, 128)
    const int NA = M_ROWS * D_EMBED;

    cvt(q, actbf, NA);
    gemm_bf16_tdm<<<gemm_grid, 256, 0, stream>>>(actbf, Wqbf, bq, Qf, M_ROWS, D_EMBED, D_EMBED);
    cvt(k, actbf, NA);
    gemm_bf16_tdm<<<gemm_grid, 256, 0, stream>>>(actbf, Wkbf, bk, Kf, M_ROWS, D_EMBED, D_EMBED);
    cvt(v, actbf, NA);
    gemm_bf16_tdm<<<gemm_grid, 256, 0, stream>>>(actbf, Wvbf, bv, Vf, M_ROWS, D_EMBED, D_EMBED);

    global_q_kernel<<<B_BATCH * H_HEADS, 256, 0, stream>>>(Qf, q_alpha, gq);
    global_v_fuse_kernel<<<B_BATCH * H_HEADS, 256, 0, stream>>>(Qf, Kf, Vf, gq, k_beta, actbf);

    gemm_bf16_tdm<<<gemm_grid, 256, 0, stream>>>(actbf, Wobf, bo, (float*)d_out, M_ROWS, D_EMBED, D_EMBED);
}
